// SpectralAdaptiveScan_46686294507587
// MI455X (gfx1250) — matrix operand order repaired and checked by reading, not compiled, not run on hardware
//
#include <hip/hip_runtime.h>
#include <hip/hip_bf16.h>
#include <math.h>

typedef __attribute__((ext_vector_type(16))) _Float16 v16h;
typedef __attribute__((ext_vector_type(8)))  float    v8f;

#define Bb 8
#define Cc 512
#define Ll 4096
#define Gg 4
#define GS 128

// Branchless exact-GELU: erf via Abramowitz–Stegun 7.1.26 (|err| < 1.5e-7).
// Output feeds an f16 A-fragment (rel. err ~5e-4), so the approximation error
// is far below quantization noise, and it removes the divergent two-path
// polynomial in libm erff from the hot loop.
__device__ __forceinline__ float gelu_erf_fast(float t) {
  float z  = t * 0.70710678118654752f;      // t / sqrt(2)
  float az = __builtin_fabsf(z);
  float u  = __builtin_amdgcn_rcpf(fmaf(0.3275911f, az, 1.0f));
  float p  = fmaf(u, 1.061405429f, -1.453152027f);
  p = fmaf(u, p, 1.421413741f);
  p = fmaf(u, p, -0.284496736f);
  p = fmaf(u, p, 0.254829592f);
  p = p * u;
  float e  = __expf(-az * az);              // v_exp_f32, branchless
  float r  = fmaf(-p, e, 1.0f);             // erf(|z|)
  float er = copysignf(r, z);
  return 0.5f * t * (1.0f + er);
}

// ---------------------------------------------------------------------------
// Pass 1: depthwise conv (k=3, pad=1) recomputed on the fly; per-(c,b) partial
// sum / sumsq for BatchNorm batch statistics. Deterministic tree reduction.
// Branchless halo: clamped address + multiplicative mask -> unconditional,
// coalesced loads.   grid = (512, 8), block = 256
// ---------------------------------------------------------------------------
__global__ void conv_stats_kernel(const float* __restrict__ x,
                                  const float* __restrict__ conv_w,
                                  const float* __restrict__ conv_b,
                                  float* __restrict__ psum,
                                  float* __restrict__ psq) {
  const int c = blockIdx.x;
  const int b = blockIdx.y;
  const int tid = threadIdx.x;
  const float w0 = conv_w[c * 3 + 0];
  const float w1 = conv_w[c * 3 + 1];
  const float w2 = conv_w[c * 3 + 2];
  const float cb = conv_b[c];
  const float* xc = x + ((size_t)b * Cc + c) * Ll;

  float s = 0.0f, q = 0.0f;
  for (int i = 0; i < Ll / 256; ++i) {
    int l = i * 256 + tid;
    int lm = (l > 0) ? l - 1 : 0;
    int lp = (l < Ll - 1) ? l + 1 : Ll - 1;
    float mm = (l > 0) ? 1.0f : 0.0f;
    float mp = (l < Ll - 1) ? 1.0f : 0.0f;
    float xm = xc[lm] * mm;
    float x0 = xc[l];
    float xp = xc[lp] * mp;
    float y = fmaf(w0, xm, fmaf(w1, x0, fmaf(w2, xp, cb)));
    s += y;
    q = fmaf(y, y, q);
  }

  __shared__ float rs[256], rq[256];
  rs[tid] = s; rq[tid] = q;
  __syncthreads();
  for (int off = 128; off > 0; off >>= 1) {
    if (tid < off) { rs[tid] += rs[tid + off]; rq[tid] += rq[tid + off]; }
    __syncthreads();
  }
  if (tid == 0) {
    psum[c * Bb + b] = rs[0];
    psq [c * Bb + b] = rq[0];
  }
}

// ---------------------------------------------------------------------------
// Pass 2: fold BN (+conv bias) into per-channel scale/shift.
// grid = 2, block = 256
// ---------------------------------------------------------------------------
__global__ void bn_finalize_kernel(const float* __restrict__ psum,
                                   const float* __restrict__ psq,
                                   const float* __restrict__ gamma,
                                   const float* __restrict__ beta,
                                   const float* __restrict__ conv_b,
                                   float* __restrict__ scp,
                                   float* __restrict__ shp) {
  int c = blockIdx.x * 256 + threadIdx.x;
  if (c >= Cc) return;
  float s = 0.0f, q = 0.0f;
  for (int b = 0; b < Bb; ++b) { s += psum[c * Bb + b]; q += psq[c * Bb + b]; }
  const float invN = 1.0f / (float)(Bb * Ll);
  float mean = s * invN;
  float var  = q * invN - mean * mean;
  float rsq  = rsqrtf(var + 1e-5f);
  float sc   = gamma[c] * rsq;
  scp[c] = sc;
  shp[c] = (conv_b[c] - mean) * sc + beta[c];
}

// ---------------------------------------------------------------------------
// Pass 3: fused conv->BN->GELU feature + WMMA projections (f16 in, f32 acc).
// A-frag (16x32 f16): lane L holds row L&15; e<8 -> K=e+8*half, e>=8 ->
// K=16+(e-8)+8*half (half = L>=16). B-frag (32x16 f16): lane L holds col
// L&15, same K mapping as A. D: VGPR r -> M=r+8*half, N=lane&15.
// B fragments for chunk ci+1 are prefetched before the chunk-ci WMMAs issue.
// grid = (L/128, B), block = 256 (8 waves)
// ---------------------------------------------------------------------------
__global__ void proj_wmma_kernel(const float* __restrict__ x,
                                 const float* __restrict__ conv_w,
                                 const float* __restrict__ scp,
                                 const float* __restrict__ shp,
                                 const float* __restrict__ off_w,
                                 const float* __restrict__ off_b,
                                 const float* __restrict__ wgt_w,
                                 const float* __restrict__ wgt_b,
                                 float* __restrict__ offs_out,
                                 float* __restrict__ wgts_out) {
  __shared__ v16h  btab[2][16][32];            // 32 KB: B fragments per K-chunk
  __shared__ float stage[8][2][16][16];        // 16 KB: D tiles for softmax pass
  __shared__ float lw0[Cc], lw1[Cc], lw2[Cc], lsc[Cc], lsh[Cc];  // 10 KB

  const int tid = threadIdx.x;
  const int b   = blockIdx.y;
  const int l0  = blockIdx.x * 128;

  for (int c = tid; c < Cc; c += 256) {
    lw0[c] = conv_w[c * 3 + 0];
    lw1[c] = conv_w[c * 3 + 1];
    lw2[c] = conv_w[c * 3 + 2];
    lsc[c] = scp[c];
    lsh[c] = shp[c];
  }
  // Build B fragments for both weight matrices (pad N: 12 -> 16 with zeros).
  for (int item = tid; item < 2 * 16 * 32; item += 256) {
    int which = item >> 9;           // 0 = offsets, 1 = weights
    int ci    = (item >> 5) & 15;    // K-chunk
    int ln    = item & 31;           // lane slot
    int n     = ln & 15;
    int khalf = (ln >= 16) ? 16 : 0;
    const float* W = which ? wgt_w : off_w;
    v16h v;
    #pragma unroll
    for (int e = 0; e < 16; ++e) {
      int Kg = ci * 32 + ((e < 8) ? (e + khalf / 2) : (16 + (e - 8) + khalf / 2));
      float val = (n < 12) ? W[n * Cc + Kg] : 0.0f;
      v[e] = (_Float16)val;
    }
    btab[which][ci][ln] = v;
  }
  __syncthreads();

  const int wave = tid >> 5;
  const int lane = tid & 31;
  const int mrow = lane & 15;
  const int hlf  = lane >> 4;
  const int lrow = l0 + wave * 16 + mrow;
  const float* xb = x + (size_t)b * Cc * Ll;

  // Branchless conv halo: clamped offsets + value masks.
  const int   lm = (lrow > 0) ? lrow - 1 : 0;
  const int   lp = (lrow < Ll - 1) ? lrow + 1 : Ll - 1;
  const float mm = (lrow > 0) ? 1.0f : 0.0f;
  const float mp = (lrow < Ll - 1) ? 1.0f : 0.0f;

  v8f acc_o = {0.f, 0.f, 0.f, 0.f, 0.f, 0.f, 0.f, 0.f};
  v8f acc_w = {0.f, 0.f, 0.f, 0.f, 0.f, 0.f, 0.f, 0.f};

  v16h bo = btab[0][0][lane];
  v16h bw = btab[1][0][lane];
  for (int ci = 0; ci < 16; ++ci) {
    v16h a;
    #pragma unroll
    for (int e = 0; e < 16; ++e) {
      int Kc = ci * 32 + ((e < 8) ? (e + hlf * 8) : (16 + (e - 8) + hlf * 8));
      const float* xc = xb + (size_t)Kc * Ll;
      float xm = xc[lm] * mm;
      float x0 = xc[lrow];
      float xp = xc[lp] * mp;
      float t = fmaf(fmaf(lw0[Kc], xm, fmaf(lw1[Kc], x0, lw2[Kc] * xp)),
                     lsc[Kc], lsh[Kc]);
      a[e] = (_Float16)gelu_erf_fast(t);
    }
    v16h bo_cur = bo, bw_cur = bw;
    int cin = (ci + 1) & 15;                  // prefetch next chunk's B frags
    bo = btab[0][cin][lane];
    bw = btab[1][cin][lane];
    acc_o = __builtin_amdgcn_wmma_f32_16x16x32_f16(false, a, false, bo_cur,
                                                   (short)0, acc_o, false, false);
    acc_w = __builtin_amdgcn_wmma_f32_16x16x32_f16(false, a, false, bw_cur,
                                                   (short)0, acc_w, false, false);
  }

  const int n = lane & 15;
  float obias = (n < 12) ? off_b[n] : 0.0f;
  float wbias = (n < 12) ? wgt_b[n] : 0.0f;
  #pragma unroll
  for (int r = 0; r < 8; ++r) {
    int m = r + hlf * 8;
    stage[wave][0][m][n] = acc_o[r] + obias;
    stage[wave][1][m][n] = acc_w[r] + wbias;
  }
  __syncthreads();

  // Per-row: softmax over K within each group; emit offsets + weights.
  if (tid < 128) {
    int w = tid >> 4, m = tid & 15;
    size_t row = (size_t)b * Ll + l0 + tid;   // block-row index == tid
    float* orow = offs_out + row * 12;
    float* wrow = wgts_out + row * 12;
    #pragma unroll
    for (int g = 0; g < Gg; ++g) {
      float z0 = stage[w][1][m][g * 3 + 0];
      float z1 = stage[w][1][m][g * 3 + 1];
      float z2 = stage[w][1][m][g * 3 + 2];
      float mx = fmaxf(z0, fmaxf(z1, z2));
      float e0 = __expf(z0 - mx), e1 = __expf(z1 - mx), e2 = __expf(z2 - mx);
      float inv = 1.0f / (e0 + e1 + e2);
      wrow[g * 3 + 0] = e0 * inv;
      wrow[g * 3 + 1] = e1 * inv;
      wrow[g * 3 + 2] = e2 * inv;
      orow[g * 3 + 0] = stage[w][0][m][g * 3 + 0];
      orow[g * 3 + 1] = stage[w][0][m][g * 3 + 1];
      orow[g * 3 + 2] = stage[w][0][m][g * 3 + 2];
    }
  }
}

// ---------------------------------------------------------------------------
// Pass 4: deformable gather + bilinear interp + softmax blend.
// Block = (b, 32-l tile). x slab staged in padded LDS [32][513] (dynamic,
// ~66 KB; fine given 320 KB LDS/WGP). Lanes map to l => coalesced global
// loads/stores; padded stride keeps the per-lane channel gathers
// conflict-free.   grid = (L/32, B), block = 256
// ---------------------------------------------------------------------------
__global__ void gather_kernel(const float* __restrict__ x,
                              const float* __restrict__ offs_in,
                              const float* __restrict__ wgts_in,
                              float* __restrict__ out) {
  extern __shared__ float slab[];              // [32][513] padded
  __shared__ float soff[32][12];
  __shared__ float swgt[32][12];

  const int tid = threadIdx.x;
  const int b   = blockIdx.y;
  const int l0  = blockIdx.x * 32;

  {
    int l   = tid & 31;
    int cb0 = tid >> 5;                        // 0..7
    const float* xb = x + (size_t)b * Cc * Ll + l0 + l;
    for (int i = 0; i < 64; ++i) {
      int c = cb0 * 64 + i;
      slab[l * 513 + c] = xb[(size_t)c * Ll];  // conflict-free: stride 513 words
    }
  }
  for (int idx = tid; idx < 2 * 32 * 12; idx += 256) {
    int which = idx / 384;
    int rem   = idx - which * 384;
    int row = rem / 12, n = rem % 12;
    size_t g = ((size_t)b * Ll + l0 + row) * 12 + n;
    if (which == 0) soff[row][n] = offs_in[g];
    else            swgt[row][n] = wgts_in[g];
  }
  __syncthreads();

  const int l  = tid & 31;
  const int jb = tid >> 5;
  float o[Gg][3], w[Gg][3];
  #pragma unroll
  for (int g = 0; g < Gg; ++g)
    #pragma unroll
    for (int k = 0; k < 3; ++k) {
      o[g][k] = soff[l][g * 3 + k];
      w[g][k] = swgt[l][g * 3 + k];
    }

  const float* srow = slab + l * 513;
  float* ob = out + (size_t)b * Cc * Ll + l0 + l;

  for (int jj = 0; jj < 16; ++jj) {
    int j = jj * 8 + jb;
    #pragma unroll
    for (int g = 0; g < Gg; ++g) {
      const float* sg = srow + g * GS;
      float acc = 0.0f;
      #pragma unroll
      for (int k = 0; k < 3; ++k) {
        float idxf = fminf(fmaxf((float)j + o[g][k], 0.0f), (float)(GS - 1));
        float fl = floorf(idxf);
        float fr = idxf - fl;
        int   fi = (int)fl;
        int   ce = (int)ceilf(idxf);
        float xf = sg[fi];
        float xc = sg[ce];
        acc = fmaf(w[g][k], fmaf(xc - xf, fr, xf), acc);
      }
      ob[(size_t)(g * GS + j) * Ll] = acc;     // coalesced over l (lanes)
    }
  }
}

// ---------------------------------------------------------------------------
// Workspace layout (floats):
//   [0,4096)        psum[c][b]
//   [4096,8192)     psq[c][b]
//   [8192,8704)     scale[c]
//   [8704,9216)     shift[c]
//   [9216, +393216) offsets (B*L*12)
//   then            weights (B*L*12)      total ~3.2 MB
// ---------------------------------------------------------------------------
extern "C" void kernel_launch(void* const* d_in, const int* in_sizes, int n_in,
                              void* d_out, int out_size, void* d_ws, size_t ws_size,
                              hipStream_t stream) {
  (void)in_sizes; (void)n_in; (void)out_size; (void)ws_size;
  const float* x        = (const float*)d_in[1];
  const float* conv_w   = (const float*)d_in[2];
  const float* conv_b   = (const float*)d_in[3];
  const float* bn_gamma = (const float*)d_in[4];
  const float* bn_beta  = (const float*)d_in[5];
  const float* off_w    = (const float*)d_in[6];
  const float* off_b    = (const float*)d_in[7];
  const float* wgt_w    = (const float*)d_in[8];
  const float* wgt_b    = (const float*)d_in[9];
  float* out = (float*)d_out;
  float* ws  = (float*)d_ws;

  float* psum = ws;
  float* psq  = ws + 4096;
  float* scp  = ws + 8192;
  float* shp  = ws + 8704;
  float* offs = ws + 9216;
  float* wgts = ws + 9216 + (size_t)Bb * Ll * 12;

  conv_stats_kernel<<<dim3(Cc, Bb), 256, 0, stream>>>(x, conv_w, conv_b, psum, psq);
  bn_finalize_kernel<<<2, 256, 0, stream>>>(psum, psq, bn_gamma, bn_beta, conv_b, scp, shp);
  proj_wmma_kernel<<<dim3(Ll / 128, Bb), 256, 0, stream>>>(
      x, conv_w, scp, shp, off_w, off_b, wgt_w, wgt_b, offs, wgts);
  gather_kernel<<<dim3(Ll / 32, Bb), 256, 32 * 513 * sizeof(float), stream>>>(
      x, offs, wgts, out);
}
